// Block_Attention_9225589752303
// MI455X (gfx1250) — matrix operand order repaired and checked by reading, not compiled, not run on hardware
//
#include <hip/hip_runtime.h>

typedef __attribute__((ext_vector_type(16))) __bf16 v16bf;
typedef __attribute__((ext_vector_type(8)))  float  v8f;
typedef __attribute__((ext_vector_type(4)))  unsigned int u32x4;

#define BB 4
#define CC 2048
#define HWD 64
#define NPIX 4096
#define NPATCH 64
#define PSQ 64
#define CK 256
#define MCAT 2560   // 256 K rows + 256 Q rows + 2048 V rows

// ---- WMMA operand loaders (layouts per CDNA5 ISA 7.12.2) -------------------
// A 16x32 bf16: lane<16 holds K = {0..7,16..23}, lane>=16 K = {8..15,24..31}
__device__ __forceinline__ v16bf ld_a(const __bf16* base, int kk0 /*0|8*/) {
  union { v16bf v; u32x4 q[2]; } u;
  u.q[0] = *(const u32x4*)(base + kk0);
  u.q[1] = *(const u32x4*)(base + kk0 + 16);
  return u.v;
}
// B 32x16 bf16: lane<16 holds K = {0..7,16..23}, lane>=16 K = {8..15,24..31} (col = lane&15)
__device__ __forceinline__ v16bf ld_b(const __bf16* base, int kk0 /*0|16*/) {
  union { v16bf v; u32x4 q[2]; } u;
  u.q[0] = *(const u32x4*)(base + kk0 / 2);
  u.q[1] = *(const u32x4*)(base + kk0 / 2 + 16);
  return u.v;
}
__device__ __forceinline__ v8f wmma_bf(v16bf a, v16bf b, v8f c) {
  return __builtin_amdgcn_wmma_f32_16x16x32_bf16(false, a, false, b, (short)0, c,
                                                 false, false);
}

// ---- 1) x[b,c,h,w] f32 -> xP[b,pix, s*32+c_lo] bf16  (s=ch%64, c_lo=ch/64) --
#define TP_PIX 8
#define TP_STRIDE 2056   // padded row to dodge LDS bank conflicts
__global__ __launch_bounds__(256) void k_transpose(const float* __restrict__ x,
                                                   __bf16* __restrict__ xP) {
  __shared__ __bf16 lds[TP_PIX * TP_STRIDE];
  const int b = blockIdx.y;
  const int pix0 = blockIdx.x * TP_PIX;
  const float* xb = x + (size_t)b * CC * NPIX;
  const int t = threadIdx.x;
  const int p = t & (TP_PIX - 1);
  const int chl = t >> 3;                       // 0..31
  for (int chb = 0; chb < CC; chb += 32) {
    int ch = chb + chl;
    float v = xb[(size_t)ch * NPIX + pix0 + p];
    lds[p * TP_STRIDE + (ch & 63) * 32 + (ch >> 6)] = (__bf16)v;
  }
  __syncthreads();
  const int pw = t >> 5;                        // 0..7
  const int lane = t & 31;
  __bf16* dst = xP + ((size_t)b * NPIX + pix0 + pw) * CC;
  const __bf16* src = lds + pw * TP_STRIDE;
  for (int it = 0; it < CC; it += 256) {
    int off = it + lane * 8;
    *(u32x4*)(dst + off) = *(const u32x4*)(src + off);
  }
}

// ---- 2) pack weights to bf16 ----------------------------------------------
__global__ __launch_bounds__(256) void k_prep(const float* __restrict__ Wk,
                                              const float* __restrict__ Wq,
                                              const float* __restrict__ Wv,
                                              const float* __restrict__ incW,
                                              __bf16* __restrict__ Wcat,
                                              __bf16* __restrict__ incWp) {
  size_t i = (size_t)blockIdx.x * 256 + threadIdx.x;
  const size_t ncat = (size_t)MCAT * CC;
  if (i < ncat) {
    int r = (int)(i >> 11), c = (int)(i & 2047);
    float v = (r < 256) ? Wk[(size_t)r * CC + c]
            : (r < 512) ? Wq[(size_t)(r - 256) * CC + c]
                        : Wv[(size_t)(r - 512) * CC + c];
    Wcat[i] = (__bf16)v;
  } else if (i < ncat + (size_t)CK * CC) {
    size_t ii = i - ncat;
    int k = (int)(ii >> 11), ch = (int)(ii & 2047);
    incWp[(size_t)k * CC + (ch & 63) * 32 + (ch >> 6)] = (__bf16)incW[ii];
  }
}

// ---- 3) incidence: inc_Wp[256,2048] @ xP_pixel -> BN -> ReLU (h<16 only) ---
__global__ __launch_bounds__(256) void k_incidence(
    const __bf16* __restrict__ xP, const __bf16* __restrict__ incWp,
    const float* __restrict__ incb, const float* __restrict__ gam,
    const float* __restrict__ bet, const float* __restrict__ mea,
    const float* __restrict__ var, float* __restrict__ incbuf) {
  const int b = blockIdx.y;
  const int pix0 = blockIdx.x * 64;               // 16 tiles * 64 = 1024 pixels
  const int wv = threadIdx.x >> 5, lane = threadIdx.x & 31;
  const int li = lane & 15, hi = lane >> 4;
  const int kk0a = hi * 8, kk0b = hi * 16;
  const int m0 = wv * 32;
  const __bf16* xb = xP + (size_t)b * NPIX * CC;
  const __bf16* a0 = incWp + (size_t)(m0 + li) * CC;
  const __bf16* a1 = a0 + 16 * CC;
  v8f acc[2][4] = {};
  for (int kb = 0; kb < CC; kb += 32) {
    v16bf A0 = ld_a(a0 + kb, kk0a);
    v16bf A1 = ld_a(a1 + kb, kk0a);
#pragma unroll
    for (int nt = 0; nt < 4; ++nt) {
      v16bf B = ld_b(xb + (size_t)(pix0 + nt * 16 + li) * CC + kb, kk0b);
      acc[0][nt] = wmma_bf(A0, B, acc[0][nt]);
      acc[1][nt] = wmma_bf(A1, B, acc[1][nt]);
    }
  }
#pragma unroll
  for (int mt = 0; mt < 2; ++mt)
#pragma unroll
    for (int nt = 0; nt < 4; ++nt)
#pragma unroll
      for (int v = 0; v < 8; ++v) {
        int k = m0 + mt * 16 + v + 8 * hi;
        float val = acc[mt][nt][v] + incb[k];
        val = (val - mea[k]) * (gam[k] / sqrtf(var[k] + 1e-5f)) + bet[k];
        val = fmaxf(val, 0.f);
        incbuf[((size_t)b * CK + k) * 1024 + pix0 + nt * 16 + li] = val;
      }
}

// ---- 4) 4x4 avg/max pool -> inc_flat[b,k,n] (n<64 => hp<4) -----------------
__global__ __launch_bounds__(256) void k_pool(const float* __restrict__ incbuf,
                                              float* __restrict__ incfl) {
  int t = blockIdx.x * 256 + threadIdx.x;           // 4*256*64
  int n = t & 63, k = (t >> 6) & 255, b = t >> 14;
  int hp = n >> 4, wp = n & 15;
  const float* p = incbuf + ((size_t)b * CK + k) * 1024;
  float s = 0.f, m = -3.4e38f;
#pragma unroll
  for (int dh = 0; dh < 4; ++dh)
#pragma unroll
    for (int dw = 0; dw < 4; ++dw) {
      float v = p[(hp * 4 + dh) * HWD + wp * 4 + dw];
      s += v; m = fmaxf(m, v);
    }
  incfl[t] = fmaxf(s * (1.f / 16.f), 0.f) + fmaxf(m, 0.f);
}

// ---- 5) big projection GEMM: Wcat[2560,2048] @ X_n[2048,64] ----------------
// Two patches per workgroup: A tiles (Wcat) are shared by both patches, which
// halves Wcat L2 traffic and raises WMMA density per load (16 WMMA / 10 loads).
__global__ __launch_bounds__(256) void k_proj(
    const __bf16* __restrict__ xP, const __bf16* __restrict__ Wcat,
    const float* __restrict__ bk, const float* __restrict__ bq,
    const float* __restrict__ bv, __bf16* __restrict__ Kt,
    float* __restrict__ Qp, __bf16* __restrict__ Vb) {
  const int b = blockIdx.z, n0 = blockIdx.y * 2, m0 = blockIdx.x * 256;
  const int h0 = (n0 >> 3) << 3;                // both patches share h0
  const int wv = threadIdx.x >> 5, lane = threadIdx.x & 31;
  const int li = lane & 15, hi = lane >> 4;
  const int kk0a = hi * 8, kk0b = hi * 16;
  const int mrow = m0 + wv * 32;
  const __bf16* a0 = Wcat + (size_t)(mrow + li) * CC;
  const __bf16* a1 = a0 + 16 * CC;
  const __bf16* xb = xP + (size_t)b * NPIX * CC;
  const int w0a = (n0 & 7) << 3;                // patch 0 w-origin
  v8f acc[2][2][4] = {};                        // [patch][mt][nt]
  for (int g = 0; g < 64; ++g) {
    const int kb = g * 32;
    const int pix0g = (h0 + (g >> 3)) * HWD + w0a + (g & 7);
    if (g + 1 < 64) {
      int pn = (h0 + ((g + 1) >> 3)) * HWD + w0a + ((g + 1) & 7);
      __builtin_prefetch(xb + (size_t)pn * CC, 0, 0);            // patch 0
      __builtin_prefetch(xb + (size_t)(pn + 8) * CC, 0, 0);      // patch 1
    }
    v16bf A0 = ld_a(a0 + kb, kk0a);
    v16bf A1 = ld_a(a1 + kb, kk0a);
#pragma unroll
    for (int pp = 0; pp < 2; ++pp) {
      const __bf16* pb = xb + (size_t)(pix0g + pp * 8) * CC;     // w0 differs by 8
#pragma unroll
      for (int nt = 0; nt < 4; ++nt) {
        v16bf B = ld_b(pb + (nt * 16 + li) * 32, kk0b);
        acc[pp][0][nt] = wmma_bf(A0, B, acc[pp][0][nt]);
        acc[pp][1][nt] = wmma_bf(A1, B, acc[pp][1][nt]);
      }
    }
  }
#pragma unroll
  for (int pp = 0; pp < 2; ++pp) {
    const int n = n0 + pp;
#pragma unroll
    for (int mt = 0; mt < 2; ++mt)
#pragma unroll
      for (int nt = 0; nt < 4; ++nt)
#pragma unroll
        for (int v = 0; v < 8; ++v) {
          int m = mrow + mt * 16 + v + 8 * hi;
          int s = nt * 16 + li;
          float val = acc[pp][mt][nt][v];
          if (m < 256) {
            Kt[(((size_t)b * NPATCH + n) * PSQ + s) * CK + m] =
                (__bf16)(val + bk[m]);
          } else if (m < 512) {
            int q = m - 256;
            Qp[(((size_t)b * NPATCH + n) * CK + q) * PSQ + s] = val + bq[q];
          } else {
            int c = m - 512;
            Vb[(((size_t)b * NPATCH + n) * CC + c) * PSQ + s] =
                (__bf16)(val + bv[c]);
          }
        }
  }
}

// ---- 6) Q_block[b,k,s] = sum_n inc_flat[b,k,n]*Q[b,n,k,s]; store transposed-
__global__ __launch_bounds__(256) void k_qblock(const float* __restrict__ Qp,
                                                const float* __restrict__ incfl,
                                                __bf16* __restrict__ QbT) {
  int t = blockIdx.x * 256 + threadIdx.x;           // 4*256*64
  int s = t & 63, k = (t >> 6) & 255, b = t >> 14;
  const float* qb = Qp + (size_t)b * NPATCH * CK * PSQ + (size_t)k * PSQ + s;
  const float* in = incfl + ((size_t)b * CK + k) * 64;
  float acc = 0.f;
#pragma unroll 4
  for (int n = 0; n < NPATCH; ++n) acc += in[n] * qb[(size_t)n * CK * PSQ];
  QbT[((size_t)b * PSQ + s) * CK + k] = (__bf16)acc;
}

// ---- 7) scores = Qb^T K, softmax over x (rows), write A^T bf16 -------------
__global__ __launch_bounds__(256) void k_scores(const __bf16* __restrict__ QbT,
                                                const __bf16* __restrict__ Kt,
                                                __bf16* __restrict__ At) {
  __shared__ float sc[64 * 64];
  const int b = blockIdx.y, n = blockIdx.x;
  const int wv = threadIdx.x >> 5, lane = threadIdx.x & 31;
  const int li = lane & 15, hi = lane >> 4;
  const int kk0a = hi * 8, kk0b = hi * 16;
  const int xt = wv >> 1, yt0 = (wv & 1) * 2;
  const __bf16* Arow = QbT + ((size_t)b * PSQ + xt * 16 + li) * CK;
  const __bf16* Bbas = Kt + ((size_t)b * NPATCH + n) * PSQ * CK;
  v8f acc[2] = {};
  for (int kb = 0; kb < CK; kb += 32) {
    v16bf A = ld_a(Arow + kb, kk0a);
#pragma unroll
    for (int u = 0; u < 2; ++u) {
      v16bf B = ld_b(Bbas + (size_t)((yt0 + u) * 16 + li) * CK + kb, kk0b);
      acc[u] = wmma_bf(A, B, acc[u]);
    }
  }
#pragma unroll
  for (int u = 0; u < 2; ++u)
#pragma unroll
    for (int v = 0; v < 8; ++v)
      sc[(xt * 16 + v + 8 * hi) * 64 + (yt0 + u) * 16 + li] = acc[u][v];
  __syncthreads();
  if (threadIdx.x < 64) {
    const int y = threadIdx.x;
    float m = -3.4e38f;
    for (int xx = 0; xx < 64; ++xx) m = fmaxf(m, sc[xx * 64 + y]);
    float ssum = 0.f;
    for (int xx = 0; xx < 64; ++xx) ssum += __expf(sc[xx * 64 + y] - m);
    float inv = 1.f / ssum;
    __bf16* dst = At + (((size_t)b * NPATCH + n) * PSQ + y) * PSQ;
    for (int xx = 0; xx < 64; ++xx)
      dst[xx] = (__bf16)(__expf(sc[xx * 64 + y] - m) * inv);
  }
}

// ---- 8) out = V @ A (+ residual), faithful final layout [b,c,h=n,w=y] ------
__global__ __launch_bounds__(256) void k_out(const __bf16* __restrict__ Vb,
                                             const __bf16* __restrict__ At,
                                             const float* __restrict__ x,
                                             float* __restrict__ out) {
  const int b = blockIdx.z, n = blockIdx.y;
  const int wv = threadIdx.x >> 5, lane = threadIdx.x & 31;
  const int li = lane & 15, hi = lane >> 4;
  const int kk0a = hi * 8, kk0b = hi * 16;
  const int c0 = blockIdx.x * 256 + wv * 32;
  const __bf16* a0 = Vb + (((size_t)b * NPATCH + n) * CC + c0 + li) * PSQ;
  const __bf16* a1 = a0 + 16 * PSQ;
  const __bf16* Bb = At + ((size_t)b * NPATCH + n) * PSQ * PSQ;
  v8f acc[2][4] = {};
#pragma unroll
  for (int kb = 0; kb < PSQ; kb += 32) {
    v16bf A0 = ld_a(a0 + kb, kk0a);
    v16bf A1 = ld_a(a1 + kb, kk0a);
#pragma unroll
    for (int nt = 0; nt < 4; ++nt) {
      v16bf B = ld_b(Bb + (size_t)(nt * 16 + li) * PSQ + kb, kk0b);
      acc[0][nt] = wmma_bf(A0, B, acc[0][nt]);
      acc[1][nt] = wmma_bf(A1, B, acc[1][nt]);
    }
  }
#pragma unroll
  for (int mt = 0; mt < 2; ++mt)
#pragma unroll
    for (int nt = 0; nt < 4; ++nt)
#pragma unroll
      for (int v = 0; v < 8; ++v) {
        int c = c0 + mt * 16 + v + 8 * hi;
        int y = nt * 16 + li;
        size_t idx = ((size_t)b * CC + c) * NPIX + (size_t)n * PSQ + y;
        out[idx] = acc[mt][nt][v] + x[idx];
      }
}

// ---------------------------------------------------------------------------
extern "C" void kernel_launch(void* const* d_in, const int* in_sizes, int n_in,
                              void* d_out, int out_size, void* d_ws, size_t ws_size,
                              hipStream_t stream) {
  (void)in_sizes; (void)n_in; (void)out_size; (void)ws_size;
  const float* x    = (const float*)d_in[0];
  const float* Wk   = (const float*)d_in[1];
  const float* bk   = (const float*)d_in[2];
  const float* Wq   = (const float*)d_in[3];
  const float* bq   = (const float*)d_in[4];
  const float* Wv   = (const float*)d_in[5];
  const float* bv   = (const float*)d_in[6];
  const float* incW = (const float*)d_in[7];
  const float* incb = (const float*)d_in[8];
  const float* gam  = (const float*)d_in[9];
  const float* bet  = (const float*)d_in[10];
  const float* mea  = (const float*)d_in[11];
  const float* var  = (const float*)d_in[12];
  float* out = (float*)d_out;

  char* w = (char*)d_ws;
  __bf16* xP    = (__bf16*)w;  w += (size_t)BB * NPIX * CC * 2;
  __bf16* Wcat  = (__bf16*)w;  w += (size_t)MCAT * CC * 2;
  __bf16* incWp = (__bf16*)w;  w += (size_t)CK * CC * 2;
  float*  incbf = (float*)w;   w += (size_t)BB * CK * 1024 * 4;
  float*  incfl = (float*)w;   w += (size_t)BB * CK * 64 * 4;
  __bf16* Kt    = (__bf16*)w;  w += (size_t)BB * NPATCH * PSQ * CK * 2;
  float*  Qp    = (float*)w;   w += (size_t)BB * NPATCH * CK * PSQ * 4;
  __bf16* QbT   = (__bf16*)w;  w += (size_t)BB * PSQ * CK * 2;
  __bf16* Vb    = (__bf16*)w;  w += (size_t)BB * NPATCH * CC * PSQ * 2;
  __bf16* At    = (__bf16*)w;  w += (size_t)BB * NPATCH * PSQ * PSQ * 2;

  k_transpose<<<dim3(NPIX / TP_PIX, BB), 256, 0, stream>>>(x, xP);
  k_prep<<<dim3((unsigned)(((size_t)MCAT * CC + (size_t)CK * CC) / 256)), 256, 0,
           stream>>>(Wk, Wq, Wv, incW, Wcat, incWp);
  k_incidence<<<dim3(16, BB), 256, 0, stream>>>(xP, incWp, incb, gam, bet, mea,
                                                var, incbf);
  k_pool<<<dim3(BB * CK * 64 / 256), 256, 0, stream>>>(incbf, incfl);
  k_proj<<<dim3(10, NPATCH / 2, BB), 256, 0, stream>>>(xP, Wcat, bk, bq, bv, Kt,
                                                       Qp, Vb);
  k_qblock<<<dim3(BB * CK * 64 / 256), 256, 0, stream>>>(Qp, incfl, QbT);
  k_scores<<<dim3(NPATCH, BB), 256, 0, stream>>>(QbT, Kt, At);
  k_out<<<dim3(8, NPATCH, BB), 256, 0, stream>>>(Vb, At, x, out);
}
